// DressedQuantumNet_81956565942810
// MI455X (gfx1250) — matrix operand order repaired and checked by reading, not compiled, not run on hardware
//
#include <hip/hip_runtime.h>

// ---------------------------------------------------------------------------
// DressedQuantumNet fused kernel for MI455X (gfx1250, wave32, WMMA).
//
// out = (G(tanh(X @ Wpre^T + b_pre) * pi/2)) @ Wpost^T + b_post
// 5-qubit circuit reduced analytically (product state + Heisenberg pullback
// of Z through the CNOT ring):
//   z_w = -sin(a_w) * cos(t_w)
//   G   = [z1z2z3z4, z0z1, z0z1z2, z0z1z2z3, z0z1z2z3z4]
//
// Memory-bound: 128 MB input streamed once. GEMM via
// v_wmma_f32_16x16x32_bf16; each wave owns 32 rows (2 C tiles sharing B).
// f32->bf16 via hardware packed converts (__builtin_convertvector).
// ---------------------------------------------------------------------------

#define NQ 5
#define K_DIM 512
#define WAVES_PER_BLOCK 8
#define THREADS_PER_BLOCK (WAVES_PER_BLOCK * 32)
#define ROWS_PER_WAVE 32
#define ROWS_PER_BLOCK (ROWS_PER_WAVE * WAVES_PER_BLOCK)
// bf16 elements per padded LDS row of w_pre: 520*2 = 1040 bytes
// (16B aligned; 1040/4 = 260 ≡ 4 (mod 64) -> lanes hit distinct banks)
#define WPRE_STRIDE 520

typedef __attribute__((ext_vector_type(16))) __bf16 v16bf;
typedef __attribute__((ext_vector_type(16))) float  v16f;
typedef __attribute__((ext_vector_type(8)))  float  v8f;

union BPack {
    unsigned int u[8];
    v16bf        v;
};

static __device__ __forceinline__ unsigned int f32_to_bf16_rne(float f) {
    unsigned int x = __float_as_uint(f);
    x = x + 0x7FFFu + ((x >> 16) & 1u);
    return x >> 16;
}

// Pack 16 f32 (A-tile element order) into v16bf via hardware packed converts.
static __device__ __forceinline__ v16bf pack_a(float4 a0, float4 a1,
                                               float4 a2, float4 a3) {
    v16f f;
    f[0]  = a0.x; f[1]  = a0.y; f[2]  = a0.z; f[3]  = a0.w;
    f[4]  = a1.x; f[5]  = a1.y; f[6]  = a1.z; f[7]  = a1.w;
    f[8]  = a2.x; f[9]  = a2.y; f[10] = a2.z; f[11] = a2.w;
    f[12] = a3.x; f[13] = a3.y; f[14] = a3.z; f[15] = a3.w;
    return __builtin_convertvector(f, v16bf);
}

__global__ __launch_bounds__(THREADS_PER_BLOCK)
void dressed_qnet_fused(const float* __restrict__ in,      // (B, 512)
                        const float* __restrict__ w_pre,   // (5, 512)
                        const float* __restrict__ b_pre,   // (5,)
                        const float* __restrict__ q_par,   // (5,)
                        const float* __restrict__ w_post,  // (2, 5)
                        const float* __restrict__ b_post,  // (2,)
                        float* __restrict__ out,           // (B, 2)
                        int B)
{
    __shared__ unsigned short sB[16 * WPRE_STRIDE];              // bf16 w_pre, rows 5..15 zero
    __shared__ float sStage[WAVES_PER_BLOCK][ROWS_PER_WAVE][6];  // C staging, stride 6 skews banks

    const int tid = threadIdx.x;

    // ---- Stage w_pre into LDS as bf16, zero-padded to a full 16-wide B tile.
    for (int idx = tid; idx < 16 * K_DIM; idx += THREADS_PER_BLOCK) {
        int r = idx >> 9;          // / 512
        int c = idx & (K_DIM - 1); // % 512
        float v = (r < NQ) ? w_pre[r * K_DIM + c] : 0.0f;
        sB[r * WPRE_STRIDE + c] = (unsigned short)f32_to_bf16_rne(v);
    }
    __syncthreads();

    const int lane = tid & 31;
    const int wave = tid >> 5;
    const int n16  = lane & 15;   // row-in-tile (A) / column (B)
    const int kh   = lane >> 4;   // K-half select, 0 or 1

    const int rowBase = blockIdx.x * ROWS_PER_BLOCK + wave * ROWS_PER_WAVE;
    int row0 = rowBase + n16;          // tile 0: rows 0..15
    int row1 = rowBase + 16 + n16;     // tile 1: rows 16..31
    if (row0 >= B) row0 = B - 1;
    if (row1 >= B) row1 = B - 1;

    // A: lane reads in[row][kc*32 + kh*8 + {0..7}] and [... + 16 + {0..7}]
    const float4* aP0 = (const float4*)(in + (size_t)row0 * K_DIM + kh * 8);
    const float4* aP1 = (const float4*)(in + (size_t)row1 * K_DIM + kh * 8);
    // B: lane reads bf16 of w_pre row n16 at K kc*32 + kh*8 + {0..7} and kc*32 + 16 + kh*8 + {0..7}
    const uint4* bP = (const uint4*)((const unsigned char*)sB +
                                     n16 * (WPRE_STRIDE * 2) + kh * 16);

    v8f acc0 = {};
    v8f acc1 = {};

    #pragma unroll 4
    for (int kc = 0; kc < K_DIM / 32; ++kc) {
        float4 a0 = aP0[kc * 8 + 0];   // K = base + 0..3
        float4 a1 = aP0[kc * 8 + 1];   // K = base + 4..7
        float4 a2 = aP0[kc * 8 + 4];   // K = base + 16..19
        float4 a3 = aP0[kc * 8 + 5];   // K = base + 20..23
        float4 c0 = aP1[kc * 8 + 0];
        float4 c1 = aP1[kc * 8 + 1];
        float4 c2 = aP1[kc * 8 + 4];
        float4 c3 = aP1[kc * 8 + 5];

        uint4 b0 = bP[kc * 4 + 0];
        uint4 b1 = bP[kc * 4 + 2];
        BPack Bm;
        Bm.u[0] = b0.x; Bm.u[1] = b0.y; Bm.u[2] = b0.z; Bm.u[3] = b0.w;
        Bm.u[4] = b1.x; Bm.u[5] = b1.y; Bm.u[6] = b1.z; Bm.u[7] = b1.w;

        v16bf A0 = pack_a(a0, a1, a2, a3);
        v16bf A1 = pack_a(c0, c1, c2, c3);

        acc0 = __builtin_amdgcn_wmma_f32_16x16x32_bf16(
            false, A0, false, Bm.v, (short)0, acc0, false, false);
        acc1 = __builtin_amdgcn_wmma_f32_16x16x32_bf16(
            false, A1, false, Bm.v, (short)0, acc1, false, false);
    }

    // ---- Stage C (columns 0..4 = pre_out) into per-wave LDS.
    // Layout: VGPR r -> M = r + 8*kh, N = n16.
    if (n16 < NQ) {
        #pragma unroll
        for (int r = 0; r < 8; ++r) {
            sStage[wave][r + 8 * kh][n16]      = acc0[r];
            sStage[wave][16 + r + 8 * kh][n16] = acc1[r];
        }
    }
    __syncthreads();

    // ---- Epilogue: one lane per row (all 32 lanes).
    {
        int orow = rowBase + lane;
        if (orow < B) {
            float z[NQ];
            #pragma unroll
            for (int w = 0; w < NQ; ++w) {
                float p = sStage[wave][lane][w] + b_pre[w];
                float a = tanhf(p) * 1.57079632679489662f;   // tanh * pi/2
                z[w] = -__sinf(a) * __cosf(q_par[w]);        // single-qubit <Z>
            }
            // Z expectations pulled back through the CNOT ring.
            float g1 = z[0] * z[1];
            float g2 = g1 * z[2];
            float g3 = g2 * z[3];
            float g4 = g3 * z[4];
            float g0 = z[1] * z[2] * z[3] * z[4];
            float g[NQ] = { g0, g1, g2, g3, g4 };

            float o0 = b_post[0];
            float o1 = b_post[1];
            #pragma unroll
            for (int w = 0; w < NQ; ++w) {
                o0 = fmaf(w_post[w],      g[w], o0);
                o1 = fmaf(w_post[NQ + w], g[w], o1);
            }
            float2 ov = make_float2(o0, o1);
            *(float2*)(out + (size_t)orow * 2) = ov;
        }
    }
}

extern "C" void kernel_launch(void* const* d_in, const int* in_sizes, int n_in,
                              void* d_out, int out_size, void* d_ws, size_t ws_size,
                              hipStream_t stream) {
    const float* in     = (const float*)d_in[0];  // (B, 512)
    const float* w_pre  = (const float*)d_in[1];  // (5, 512)
    const float* b_pre  = (const float*)d_in[2];  // (5,)
    const float* q_par  = (const float*)d_in[3];  // (5,)
    const float* w_post = (const float*)d_in[4];  // (2, 5)
    const float* b_post = (const float*)d_in[5];  // (2,)
    float* out = (float*)d_out;                   // (B, 2)

    const int B = in_sizes[0] / K_DIM;
    const int nBlocks = (B + ROWS_PER_BLOCK - 1) / ROWS_PER_BLOCK;

    dressed_qnet_fused<<<nBlocks, THREADS_PER_BLOCK, 0, stream>>>(
        in, w_pre, b_pre, q_par, w_post, b_post, out, B);
}
